// LocalDotProductAttention_50070728737613
// MI455X (gfx1250) — matrix operand order repaired and checked by reading, not compiled, not run on hardware
//
#include <hip/hip_runtime.h>
#include <math.h>

#define B_    4
#define S_    4096
#define D_    128
#define WIN_  12
#define NEG_INF_ (-1000000000.0f)
#define SCALE_   0.08838834764831845f   // 1/sqrt(128)
#define NIT_  (S_ / 32)                 // 128 key-tile iterations

typedef __bf16 v16bf __attribute__((ext_vector_type(16)));
typedef __bf16 v8bf  __attribute__((ext_vector_type(8)));
typedef float  v8f   __attribute__((ext_vector_type(8)));
typedef int    v4i_a __attribute__((vector_size(16)));

#define KT_STRIDE 136   // bf16 per key row: 128 + 8 pad (272B, 16B aligned)
#define VT_STRIDE 40    // bf16 per d row:   32 + 8 pad  (80B, 16B aligned)

__device__ __forceinline__ __bf16 f2bf(float f) { return (__bf16)f; }
__device__ __forceinline__ unsigned short bf_u16(float f) {
    return __builtin_bit_cast(unsigned short, (__bf16)f);
}

// ---- CDNA5 async global->LDS copy (ASYNCcnt-tracked), 16B per lane --------
__device__ __forceinline__ void async_b128(const void* g, void* l) {
#if __has_builtin(__builtin_amdgcn_global_load_async_to_lds_b128)
    __builtin_amdgcn_global_load_async_to_lds_b128(
        (__attribute__((address_space(1))) v4i_a*)(g),
        (__attribute__((address_space(3))) v4i_a*)(l), 0, 0);
#else
    asm volatile("global_load_async_to_lds_b128 %0, %1, off"
                 :: "v"((unsigned int)(size_t)l), "v"(g) : "memory");
#endif
}
__device__ __forceinline__ void wait_asynccnt0() {
#if __has_builtin(__builtin_amdgcn_s_wait_asynccnt)
    __builtin_amdgcn_s_wait_asynccnt(0);
#else
    asm volatile("s_wait_asynccnt 0" ::: "memory");
#endif
}

// ---------------------------------------------------------------------------
// Pre-pass: K -> bf16 row-major, V -> bf16 transposed [b][d][s] in workspace.
// ---------------------------------------------------------------------------
__global__ __launch_bounds__(256) void convert_k_bf(
    const float* __restrict__ src, unsigned short* __restrict__ dst)
{
    const size_t i = ((size_t)blockIdx.x * 256 + threadIdx.x) * 8;
    float4 a = *(const float4*)(src + i);
    float4 b = *(const float4*)(src + i + 4);
    v8bf o;
    o[0]=f2bf(a.x); o[1]=f2bf(a.y); o[2]=f2bf(a.z); o[3]=f2bf(a.w);
    o[4]=f2bf(b.x); o[5]=f2bf(b.y); o[6]=f2bf(b.z); o[7]=f2bf(b.w);
    *(v8bf*)(dst + i) = o;
}

__global__ __launch_bounds__(256) void convert_v_t(
    const float* __restrict__ v, unsigned short* __restrict__ vt)
{
    __shared__ unsigned short tile[32][36];
    const int b  = blockIdx.z;
    const int d0 = blockIdx.y * 32;
    const int s0 = blockIdx.x * 32;
    const int r  = threadIdx.x >> 3;        // 0..31
    const int c  = (threadIdx.x & 7) * 4;   // 0..28
    float4 f = *(const float4*)(v + (((size_t)b * S_ + s0 + r) * D_ + d0 + c));
    tile[c + 0][r] = bf_u16(f.x);
    tile[c + 1][r] = bf_u16(f.y);
    tile[c + 2][r] = bf_u16(f.z);
    tile[c + 3][r] = bf_u16(f.w);
    __syncthreads();
    ushort4 o;
    o.x = tile[r][c + 0]; o.y = tile[r][c + 1];
    o.z = tile[r][c + 2]; o.w = tile[r][c + 3];
    *(ushort4*)(vt + (((size_t)b * D_ + d0 + r) * S_ + s0 + c)) = o;
}

// ---------------------------------------------------------------------------
// Kernel 1: global flash attention (online softmax), writes out = (O/l)*gate
// Grid: (S/128, B), block 256 (8 waves). Each wave owns a 16-row query tile.
// Computes S^T = K*Q^T so softmax stats are per-lane and P lands directly in
// A-layout for the P*V WMMA. K/V tiles double-buffer through LDS; when PRE,
// staging is pure async global->LDS copies of pre-converted bf16 data.
// ---------------------------------------------------------------------------
template <bool PRE>
__global__ __launch_bounds__(256) void fa_global_kernel(
    const float* __restrict__ q, const float* __restrict__ k,
    const float* __restrict__ v, const float* __restrict__ gw,
    const unsigned char* __restrict__ mask, float* __restrict__ out,
    const unsigned short* __restrict__ kbf, const unsigned short* __restrict__ vtw)
{
    __shared__ __align__(16) unsigned short Kt[2][32 * KT_STRIDE];
    __shared__ __align__(16) unsigned short Vt[2][128 * VT_STRIDE];
    __shared__ __align__(16) float Bias[S_];   // additive key-mask bias

    const int tid   = threadIdx.x;
    const int lane  = tid & 31;
    const int wid   = tid >> 5;
    const int b     = blockIdx.y;
    const int qbase = blockIdx.x * 128 + wid * 16;

    const int ln    = lane & 15;   // N index (q-row for S^T tiles / d for O tiles)
    const int half  = lane >> 4;
    const int khalf = half * 8;    // A-layout K offset for this lane group
    const int kgrp  = half * 16;   // K offset, contiguous order k = 16h + s (K and Q operands)

    const float* kb_p = k + (size_t)b * S_ * D_;
    const float* vb_p = v + (size_t)b * S_ * D_;
    const unsigned short* kbf_b = kbf + (size_t)b * S_ * D_;
    const unsigned short* vtw_b = vtw + (size_t)b * D_ * S_;
    const unsigned char* mrow = mask + (size_t)b * S_;

    // ---- prologue: mask -> additive bias table (0 or NEG_INF), 16 keys/thread
    {
        uint4 mm = *(const uint4*)(mrow + tid * 16);
        const unsigned int mw[4] = {mm.x, mm.y, mm.z, mm.w};
        #pragma unroll
        for (int w = 0; w < 4; ++w)
            #pragma unroll
            for (int j = 0; j < 4; ++j)
                Bias[tid * 16 + 4 * w + j] =
                    ((mw[w] >> (8 * j)) & 0xFFu) ? 0.0f : NEG_INF_;
    }

    // ---- Q in B-layout registers (lane = q-row ln; 16 consecutive d per chunk)
    // plus the gate dot-product accumulated in f32 while we have the raw data.
    const float* qrow = q + ((size_t)b * S_ + (qbase + ln)) * D_;
    v16bf qb[4];
    float gpart = 0.0f;
    #pragma unroll
    for (int c = 0; c < 4; ++c) {
        const int d0 = 32 * c + kgrp;
        #pragma unroll
        for (int e = 0; e < 16; e += 4) {
            float4 qf = *(const float4*)(qrow + d0 + e);
            float4 gf = *(const float4*)(gw + d0 + e);
            qb[c][e + 0] = f2bf(qf.x);
            qb[c][e + 1] = f2bf(qf.y);
            qb[c][e + 2] = f2bf(qf.z);
            qb[c][e + 3] = f2bf(qf.w);
            gpart += qf.x * gf.x + qf.y * gf.y + qf.z * gf.z + qf.w * gf.w;
        }
    }
    gpart += __shfl_xor(gpart, 16, 32);        // lanes ln and ln+16 hold halves
    const float gate = 1.0f / (1.0f + __expf(-gpart));

    // ---- accumulators
    v8f O[8];
    #pragma unroll
    for (int n = 0; n < 8; ++n)
        #pragma unroll
        for (int r = 0; r < 8; ++r) O[n][r] = 0.0f;
    float m_run = -3.0e38f, l_run = 0.0f;

    // staging work split (256 threads)
    const int skey = tid >> 3;         // K: key row 0..31
    const int sd0  = (tid & 7) * 16;   // K: d segment 0..112 (bf16 units)
    const int vkp  = tid & 15;         // V(cvt): key pair -> keys 2vkp, 2vkp+1
    const int vd0  = (tid >> 4) * 8;   // V(cvt): d segment 0..120
    const int avd  = tid >> 1;         // V(async): d row 0..127
    const int avh  = (tid & 1) * 16;   // V(async): key-half 0/16

    // ---- staging: K tile row-major [key][d], V tile transposed [d][key]
    auto stageKV = [&](int buf, int kb) {
        if constexpr (PRE) {
            // pure async byte movement of pre-converted bf16
            const unsigned short* gk = kbf_b + (size_t)(kb + skey) * D_ + sd0;
            unsigned short* lk = Kt[buf] + skey * KT_STRIDE + sd0;
            __builtin_prefetch(gk + 64 * D_, 0, 1);
            async_b128(gk, lk);
            async_b128(gk + 8, lk + 8);
            const unsigned short* gv = vtw_b + (size_t)avd * S_ + kb + avh;
            unsigned short* lv = Vt[buf] + avd * VT_STRIDE + avh;
            __builtin_prefetch(gv + 64, 0, 1);
            async_b128(gv, lv);
            async_b128(gv + 8, lv + 8);
        } else {
            // fallback: load f32, convert in-register, store to LDS
            const float* sk = kb_p + (size_t)(kb + skey) * D_ + sd0;
            __builtin_prefetch(sk + 64 * D_, 0, 1);
            unsigned short* dk = Kt[buf] + skey * KT_STRIDE + sd0;
            float4 f0 = *(const float4*)(sk + 0);
            float4 f1 = *(const float4*)(sk + 4);
            float4 f2 = *(const float4*)(sk + 8);
            float4 f3 = *(const float4*)(sk + 12);
            v8bf klo, khi;
            klo[0]=f2bf(f0.x); klo[1]=f2bf(f0.y); klo[2]=f2bf(f0.z); klo[3]=f2bf(f0.w);
            klo[4]=f2bf(f1.x); klo[5]=f2bf(f1.y); klo[6]=f2bf(f1.z); klo[7]=f2bf(f1.w);
            khi[0]=f2bf(f2.x); khi[1]=f2bf(f2.y); khi[2]=f2bf(f2.z); khi[3]=f2bf(f2.w);
            khi[4]=f2bf(f3.x); khi[5]=f2bf(f3.y); khi[6]=f2bf(f3.z); khi[7]=f2bf(f3.w);
            *(v8bf*)(dk + 0) = klo;
            *(v8bf*)(dk + 8) = khi;

            const float* sv0 = vb_p + (size_t)(kb + 2 * vkp) * D_ + vd0;
            const float* sv1 = sv0 + D_;
            __builtin_prefetch(sv0 + 64 * D_, 0, 1);
            float4 a0 = *(const float4*)(sv0 + 0);
            float4 a1 = *(const float4*)(sv0 + 4);
            float4 b0 = *(const float4*)(sv1 + 0);
            float4 b1 = *(const float4*)(sv1 + 4);
            unsigned short* dvb = Vt[buf] + vd0 * VT_STRIDE + 2 * vkp;
            const float av[8] = {a0.x, a0.y, a0.z, a0.w, a1.x, a1.y, a1.z, a1.w};
            const float bv[8] = {b0.x, b0.y, b0.z, b0.w, b1.x, b1.y, b1.z, b1.w};
            #pragma unroll
            for (int e = 0; e < 8; ++e) {
                unsigned int pk = (unsigned int)bf_u16(av[e]) |
                                  ((unsigned int)bf_u16(bv[e]) << 16);
                *(unsigned int*)(dvb + e * VT_STRIDE) = pk;
            }
        }
    };

    stageKV(0, 0);

    for (int it = 0; it < NIT_; ++it) {
        const int kb  = it * 32;
        const int buf = it & 1;
        if constexpr (PRE) wait_asynccnt0();
        __syncthreads();
        if (it + 1 < NIT_) stageKV(buf ^ 1, kb + 32);

        // ---- S^T tiles: D = Ktile(16x32 chunk) x Q^T(32x16), 4 d-chunks each
        v8f st[2];
        #pragma unroll
        for (int t = 0; t < 2; ++t) {
            v8f acc;
            #pragma unroll
            for (int r = 0; r < 8; ++r) acc[r] = 0.0f;
            const unsigned short* krow = Kt[buf] + (t * 16 + ln) * KT_STRIDE;
            #pragma unroll
            for (int c = 0; c < 4; ++c) {
                v8bf lo = *(const v8bf*)(krow + 32 * c + kgrp);
                v8bf hi = *(const v8bf*)(krow + 32 * c + kgrp + 8);
                v16bf a = __builtin_shufflevector(lo, hi, 0,1,2,3,4,5,6,7,8,9,10,11,12,13,14,15);
                acc = __builtin_amdgcn_wmma_f32_16x16x32_bf16(
                          false, a, false, qb[c], (short)0, acc, false, false);
            }
            st[t] = acc;
        }

        // ---- online softmax update (per-lane: this lane's q-row is ln)
        const float* bp = Bias + kb;
        float4 ba0 = *(const float4*)(bp + khalf);
        float4 ba1 = *(const float4*)(bp + khalf + 4);
        float4 bb0 = *(const float4*)(bp + 16 + khalf);
        float4 bb1 = *(const float4*)(bp + 16 + khalf + 4);
        const float bias0[8] = {ba0.x, ba0.y, ba0.z, ba0.w, ba1.x, ba1.y, ba1.z, ba1.w};
        const float bias1[8] = {bb0.x, bb0.y, bb0.z, bb0.w, bb1.x, bb1.y, bb1.z, bb1.w};
        float s0[8], s1[8];
        #pragma unroll
        for (int r = 0; r < 8; ++r) {
            s0[r] = st[0][r] * SCALE_ + bias0[r];
            s1[r] = st[1][r] * SCALE_ + bias1[r];
        }
        float tmax = -3.0e38f;
        #pragma unroll
        for (int r = 0; r < 8; ++r) tmax = fmaxf(tmax, fmaxf(s0[r], s1[r]));
        tmax = fmaxf(tmax, __shfl_xor(tmax, 16, 32));
        const float m_new = fmaxf(m_run, tmax);
        const bool  renorm = (m_new > m_run);
        const float corr = __expf(m_run - m_new);
        float rs = 0.0f;
        float p0[8], p1[8];
        #pragma unroll
        for (int r = 0; r < 8; ++r) {
            p0[r] = __expf(s0[r] - m_new);
            p1[r] = __expf(s1[r] - m_new);
            rs += p0[r] + p1[r];
        }
        rs += __shfl_xor(rs, 16, 32);
        l_run = l_run * corr + rs;
        m_run = m_new;

        // ---- P is already in A-layout (lane = q-row, keys across elements)
        v16bf pa;
        #pragma unroll
        for (int e = 0; e < 8; ++e) {
            pa[e]     = f2bf(p0[e]);
            pa[8 + e] = f2bf(p1[e]);
        }

        // rescale O only when some row's max actually moved (wave-uniform test)
        if (__any(renorm)) {
            float cj[8];
            #pragma unroll
            for (int r = 0; r < 8; ++r) cj[r] = __shfl(corr, r + khalf, 32);
            #pragma unroll
            for (int n = 0; n < 8; ++n)
                #pragma unroll
                for (int r = 0; r < 8; ++r) O[n][r] *= cj[r];
        }

        // ---- O[n] += P x V_chunk   (8 d-chunks of 16)
        #pragma unroll
        for (int n = 0; n < 8; ++n) {
            const unsigned short* vr = Vt[buf] + (16 * n + ln) * VT_STRIDE + khalf;
            v8bf lo = *(const v8bf*)(vr);
            v8bf hi = *(const v8bf*)(vr + 16);
            v16bf bvv = __builtin_shufflevector(lo, hi, 0,1,2,3,4,5,6,7,8,9,10,11,12,13,14,15);
            O[n] = __builtin_amdgcn_wmma_f32_16x16x32_bf16(
                       false, pa, false, bvv, (short)0, O[n], false, false);
        }
    }

    // ---- epilogue: out[j][d] = (O/l) * gate(j)
    const float wlane = gate / l_run;        // this lane's q-row (= ln) factor
    float wj[8];
    #pragma unroll
    for (int r = 0; r < 8; ++r) wj[r] = __shfl(wlane, r + khalf, 32);
    float* orow0 = out + ((size_t)b * S_ + qbase) * D_;
    #pragma unroll
    for (int n = 0; n < 8; ++n)
        #pragma unroll
        for (int r = 0; r < 8; ++r)
            orow0[(size_t)(r + khalf) * D_ + 16 * n + ln] = O[n][r] * wj[r];
}

// ---------------------------------------------------------------------------
// Kernel 2: local windowed attention (25 keys) + gated combine.
// One wave per query row; lane l owns dims [4l, 4l+3].
// out += local_out * (1 - gate)   (kernel 1 already wrote global_out * gate)
// ---------------------------------------------------------------------------
__global__ __launch_bounds__(256) void local_gate_kernel(
    const float* __restrict__ q, const float* __restrict__ k,
    const float* __restrict__ v, const float* __restrict__ gw,
    const unsigned char* __restrict__ mask, float* __restrict__ out)
{
    const int lane = threadIdx.x & 31;
    const int wid  = threadIdx.x >> 5;
    const int gid  = blockIdx.x * 8 + wid;      // b*S + i   (S == 4096)
    const int b    = gid >> 12;
    const int i    = gid & (S_ - 1);

    const float4 qv = *(const float4*)(q + (size_t)gid * D_ + lane * 4);

    // scores: lane jj ends up holding score for offset jj
    float sc = NEG_INF_;
    #pragma unroll
    for (int jj = 0; jj < 2 * WIN_ + 1; ++jj) {
        const int j = i - WIN_ + jj;
        const bool valid = (j >= 0) && (j < S_);
        const int jc = valid ? j : i;
        const float4 kv = *(const float4*)(k + ((size_t)b * S_ + jc) * D_ + lane * 4);
        float d = qv.x * kv.x + qv.y * kv.y + qv.z * kv.z + qv.w * kv.w;
        d += __shfl_xor(d, 16, 32); d += __shfl_xor(d, 8, 32);
        d += __shfl_xor(d, 4, 32);  d += __shfl_xor(d, 2, 32);
        d += __shfl_xor(d, 1, 32);
        const bool ok = valid && (mask[(size_t)b * S_ + jc] != 0);
        const float s = ok ? d * SCALE_ : NEG_INF_;
        if (lane == jj) sc = s;
    }
    float t = (lane < 2 * WIN_ + 1) ? sc : NEG_INF_;
    float mx = t;
    mx = fmaxf(mx, __shfl_xor(mx, 16, 32)); mx = fmaxf(mx, __shfl_xor(mx, 8, 32));
    mx = fmaxf(mx, __shfl_xor(mx, 4, 32));  mx = fmaxf(mx, __shfl_xor(mx, 2, 32));
    mx = fmaxf(mx, __shfl_xor(mx, 1, 32));
    float p = (lane < 2 * WIN_ + 1) ? __expf(t - mx) : 0.0f;
    float sum = p;
    sum += __shfl_xor(sum, 16, 32); sum += __shfl_xor(sum, 8, 32);
    sum += __shfl_xor(sum, 4, 32);  sum += __shfl_xor(sum, 2, 32);
    sum += __shfl_xor(sum, 1, 32);
    p /= sum;

    float acc0 = 0.f, acc1 = 0.f, acc2 = 0.f, acc3 = 0.f;
    #pragma unroll
    for (int jj = 0; jj < 2 * WIN_ + 1; ++jj) {
        const int j = i - WIN_ + jj;
        if (j < 0 || j >= S_) continue;          // uniform across the wave
        const float pj = __shfl(p, jj, 32);
        const float4 vv = *(const float4*)(v + ((size_t)b * S_ + j) * D_ + lane * 4);
        acc0 += pj * vv.x; acc1 += pj * vv.y; acc2 += pj * vv.z; acc3 += pj * vv.w;
    }

    // gate
    const float4 gv = *(const float4*)(gw + lane * 4);
    float g = qv.x * gv.x + qv.y * gv.y + qv.z * gv.z + qv.w * gv.w;
    g += __shfl_xor(g, 16, 32); g += __shfl_xor(g, 8, 32);
    g += __shfl_xor(g, 4, 32);  g += __shfl_xor(g, 2, 32);
    g += __shfl_xor(g, 1, 32);
    const float one_m_gate = 1.0f - 1.0f / (1.0f + __expf(-g));

    float* orow = out + (size_t)gid * D_ + lane * 4;
    float4 cur = *(float4*)orow;
    cur.x += acc0 * one_m_gate; cur.y += acc1 * one_m_gate;
    cur.z += acc2 * one_m_gate; cur.w += acc3 * one_m_gate;
    *(float4*)orow = cur;
}

extern "C" void kernel_launch(void* const* d_in, const int* in_sizes, int n_in,
                              void* d_out, int out_size, void* d_ws, size_t ws_size,
                              hipStream_t stream) {
    const float* q  = (const float*)d_in[0];
    const float* k  = (const float*)d_in[1];
    const float* v  = (const float*)d_in[2];
    const float* gw = (const float*)d_in[3];
    const unsigned char* mask = (const unsigned char*)d_in[4];
    float* out = (float*)d_out;

    const size_t nelem = (size_t)B_ * S_ * D_;
    unsigned short* kws = (unsigned short*)d_ws;
    unsigned short* vws = kws + nelem;
    const bool pre = (ws_size >= 2 * nelem * sizeof(unsigned short));

    dim3 gridA(S_ / 128, B_);
    if (pre) {
        convert_k_bf<<<dim3((unsigned)(nelem / 2048)), 256, 0, stream>>>(k, kws);
        convert_v_t<<<dim3(S_ / 32, D_ / 32, B_), 256, 0, stream>>>(v, vws);
        fa_global_kernel<true><<<gridA, 256, 0, stream>>>(q, k, v, gw, mask, out, kws, vws);
    } else {
        fa_global_kernel<false><<<gridA, 256, 0, stream>>>(q, k, v, gw, mask, out, kws, vws);
    }

    dim3 gridB(B_ * S_ / 8);
    local_gate_kernel<<<gridB, 256, 0, stream>>>(q, k, v, gw, mask, out);
}
